// MaskedHead_78520592105655
// MI455X (gfx1250) — matrix operand order repaired and checked by reading, not compiled, not run on hardware
//
#include <hip/hip_runtime.h>
#include <hip/hip_bf16.h>

// ---------------------------------------------------------------------------
// MaskedHead (single-head causal attention), MI455X / gfx1250, wave32.
//   D=1024, H=64, B=4, S=2048.
// ~5.4 GFLOP vs ~35 MB HBM traffic -> fp32-VALU compute-bound, so all GEMMs run
// on v_wmma_f32_16x16x32_f16 (f32 accumulate) in a flash-attention fusion.
// CDNA5-specific data movement: operand tiles (weight fragments, K/V chunks)
// are staged into LDS by the Tensor Data Mover (tensor_load_to_lds, TENSORcnt)
// double-buffered so the DMA of chunk k+1 overlaps the WMMAs of chunk k.
// launch_bounds(...,1) relaxes the VGPR cap so per-chunk B fragments can be
// batch-loaded into registers and ds waits graduate instead of serializing.
// Workspace (halfs): w fragments (384KB) | Qh (1MB) | Kh (1MB) | Vt (1MB).
// ---------------------------------------------------------------------------

typedef __attribute__((ext_vector_type(16))) _Float16 v16h;
typedef __attribute__((ext_vector_type(8)))  _Float16 v8h;
typedef __attribute__((ext_vector_type(8)))  float    v8f;
typedef __attribute__((ext_vector_type(4)))  float    v4f;
typedef __attribute__((ext_vector_type(4)))  unsigned int v4u;
typedef __attribute__((ext_vector_type(8)))  int      v8i;
typedef __attribute__((ext_vector_type(4)))  int      v4i;

#define D_MODEL 1024
#define HEAD    64
#define BATCH   4
#define SEQ     2048
#define NROWS   (BATCH*SEQ)              // 8192
#define WF_HALFS (32*12*32*16)           // 196608: [kchunk][mat*4+nt][lane][e]
#define WCHUNK_HALFS (12*32*16)          // 6144 halfs = 12KB per k-chunk
#define QH_HALFS (NROWS*HEAD)            // 524288
#define QH_OFF  ((size_t)WF_HALFS)
#define KH_OFF  (QH_OFF + QH_HALFS)
#define VT_OFF  (KH_OFF + QH_HALFS)

union F16x16 { v16h v; v8h h[2]; };

// --- LDS byte address: generic ptr = {aperture_hi, lds_offset} -> low 32b ---
__device__ __forceinline__ unsigned lds_addr_of(void* p) {
    return (unsigned)(unsigned long long)p;
}

// ---------------------------------------------------------------------------
// TDM: issue TENSOR_LOAD_TO_LDS for a 2D tile of dwords.
//   tile_x_dw dwords per row, tile_y rows, row stride stride_dw dwords.
// D# built per CDNA5 ISA 8.3/8.4: group0 = {count=1, lds_addr, global_addr,
// type=2}; group1 = {data_size=4B, tensor dims >= tile, tile dims, dim0
// stride}; groups 2/3 zero (<=2D tensor, higher tile dims unused).
// ---------------------------------------------------------------------------
__device__ __forceinline__ void tdm_load_2d(void* lds_ptr, const void* gptr,
                                            unsigned tile_x_dw, unsigned tile_y,
                                            unsigned stride_dw) {
    unsigned long long ga = (unsigned long long)gptr;
    unsigned lo = lds_addr_of(lds_ptr);
    v4u g0 = { 1u,                                   // count=1 (valid, user mode)
               lo,                                   // lds_addr [63:32]
               (unsigned)ga,                         // global_addr lo
               (unsigned)((ga >> 32) & 0x1ffffffu) | (2u << 30) }; // type=2
    unsigned td0 = (stride_dw > tile_x_dw) ? stride_dw : tile_x_dw; // tensor_dim0
    unsigned td1 = tile_y;                                          // tensor_dim1
    v8i g1 = {
        (int)(2u << 16),                                          // wg_mask=0, data_size=4B
        (int)((td0 & 0xffffu) << 16),                             // [63:48]=td0.lo16
        (int)(((td0 >> 16) & 0xffffu) | ((td1 & 0xffffu) << 16)), // td0.hi16 | td1.lo16
        (int)(((td1 >> 16) & 0xffffu) | (tile_x_dw << 16)),       // td1.hi16 | tile_dim0
        (int)(tile_y & 0xffffu),                                  // tile_dim1; tile_dim2=0
        (int)stride_dw,                                           // tensor_dim0_stride lo32
        0, 0
    };
    v4i gz = {0, 0, 0, 0};
#if defined(__clang_major__) && (__clang_major__ >= 23)
    v8i gz8 = {0, 0, 0, 0, 0, 0, 0, 0};
    __builtin_amdgcn_tensor_load_to_lds(g0, g1, gz, gz, gz8, 0);
#else
    __builtin_amdgcn_tensor_load_to_lds(g0, g1, gz, gz, 0);
#endif
}

// ---------------------------------------------------------------------------
// Kernel 0: repack w_{q,k,v} (f32 [1024,64]) into f16 WMMA B-fragment order,
// chunk-major so each 32-deep K chunk is one contiguous 12KB block for TDM:
// flat[((kc*12 + mat*4+nt)*32 + lane)*16 + e] = w[K*64 + col],
//   col = nt*16 + lane%16,  K = kc*32 + (lane/16)*8 + (e%8) + (e/8)*16
// (dense 16-bit B layout: lanes 0-15 hold K 0..7,16..23, lanes 16-31 hold K 8..15,24..31)
// ---------------------------------------------------------------------------
__global__ void prep_w_kernel(const float* __restrict__ wq,
                              const float* __restrict__ wk,
                              const float* __restrict__ wv,
                              _Float16* __restrict__ wf) {
    int tid = blockIdx.x * 256 + threadIdx.x;
    if (tid >= WF_HALFS) return;
    int e    = tid & 15;
    int lane = (tid >> 4) & 31;
    int f12  = tid >> 9;
    int kc   = f12 / 12;
    int f    = f12 - kc * 12;           // mat*4 + nt
    int mat  = f >> 2;
    int nt   = f & 3;
    int col = nt * 16 + (lane & 15);
    int K   = kc * 32 + ((lane >> 4) << 3) + (e & 7) + ((e >> 3) << 4);
    const float* w = (mat == 0) ? wq : (mat == 1) ? wk : wv;
    wf[tid] = (_Float16)w[K * HEAD + col];
}

// ---------------------------------------------------------------------------
// Kernel 1: QKV projection. Block = 4 waves, 16 rows/wave. Weight fragments
// for each 32-deep K chunk are TDM-loaded into a 2x12KB LDS double buffer
// (wave 0 issues, s_wait_tensorcnt + barrier publishes); x rows stream from
// global, cast f32->f16 in registers per the 16-bit A layout. All 12 B
// fragments are pulled from LDS into registers first, then the 12-deep WMMA
// chain runs. Q gets bias then *sqrt(64)=8 folded before f16; V stored
// transposed [B][64][S].
// ---------------------------------------------------------------------------
__global__ __launch_bounds__(128, 1) void qkv_kernel(
    const float* __restrict__ x,
    const float* __restrict__ bq, const float* __restrict__ bk,
    const float* __restrict__ bv,
    const _Float16* __restrict__ wf,
    _Float16* __restrict__ qh, _Float16* __restrict__ kh,
    _Float16* __restrict__ vt) {
    __shared__ __align__(32) _Float16 wbuf[2][WCHUNK_HALFS];   // 24KB
    const int lane = threadIdx.x & 31;
    const int wave = threadIdx.x >> 5;
    const int m    = lane & 15;
    const int kh2  = lane >> 4;
    const int rowbase = blockIdx.x * 64 + wave * 16;

    const v8f zero = {0.f,0.f,0.f,0.f,0.f,0.f,0.f,0.f};
    v8f acc[3][4];
#pragma unroll
    for (int mt = 0; mt < 3; ++mt)
#pragma unroll
        for (int nt = 0; nt < 4; ++nt) acc[mt][nt] = zero;

    const float* xr = x + (size_t)(rowbase + m) * D_MODEL;

    // ---- prologue: TDM-stage weight chunk 0 ----
    if (wave == 0)
        tdm_load_2d(&wbuf[0][0], wf, WCHUNK_HALFS / 2, 1, WCHUNK_HALFS / 2);
    __builtin_amdgcn_s_wait_tensorcnt(0);      // no-op for non-issuing waves
    __syncthreads();

    for (int kc = 0; kc < 32; ++kc) {
        const int cur = kc & 1;
        if (wave == 0 && kc + 1 < 32)       // prefetch next chunk via TDM
            tdm_load_2d(&wbuf[cur ^ 1][0], wf + (size_t)(kc + 1) * WCHUNK_HALFS,
                        WCHUNK_HALFS / 2, 1, WCHUNK_HALFS / 2);

        const int K0 = kc * 32;
        // ---- A fragment: cast 32 f32 -> v16h in ISA element order ----
        F16x16 a;
        v4f x0 = *(const v4f*)(xr + K0 + kh2 * 8);
        v4f x1 = *(const v4f*)(xr + K0 + kh2 * 8 + 4);
        v4f x2 = *(const v4f*)(xr + K0 + 16 + kh2 * 8);
        v4f x3 = *(const v4f*)(xr + K0 + 16 + kh2 * 8 + 4);
#pragma unroll
        for (int i = 0; i < 4; ++i) {
            a.v[i]      = (_Float16)x0[i];
            a.v[4 + i]  = (_Float16)x1[i];
            a.v[8 + i]  = (_Float16)x2[i];
            a.v[12 + i] = (_Float16)x3[i];
        }
        // ---- batch-load all 12 B fragments, then the WMMA chain ----
        v16h bf[12];
#pragma unroll
        for (int f = 0; f < 12; ++f)
            bf[f] = *(const v16h*)&wbuf[cur][(f * 32 + lane) * 16];
#pragma unroll
        for (int mt = 0; mt < 3; ++mt)
#pragma unroll
            for (int nt = 0; nt < 4; ++nt)
                acc[mt][nt] = __builtin_amdgcn_wmma_f32_16x16x32_f16(
                    false, a.v, false, bf[mt * 4 + nt], (short)0,
                    acc[mt][nt], false, false);

        __builtin_amdgcn_s_wait_tensorcnt(0);
        __syncthreads();                    // publish next buffer
    }

    // ---- epilogue: bias add in C layout (col n = m, row = r + 8*kh2) ----
#pragma unroll
    for (int nt = 0; nt < 4; ++nt) {
        const float bqv = bq[nt * 16 + m];
        const float bkv = bk[nt * 16 + m];
        const float bvv = bv[nt * 16 + m];
#pragma unroll
        for (int r = 0; r < 8; ++r) {
            const int row = rowbase + r + 8 * kh2;
            qh[(size_t)row * HEAD + nt * 16 + m] = (_Float16)((acc[0][nt][r] + bqv) * 8.0f);
            kh[(size_t)row * HEAD + nt * 16 + m] = (_Float16)(acc[1][nt][r] + bkv);
        }
        // V transposed [B][64][S]: lane owns one d, 8 consecutive s -> b128 store
        v8h vpack;
#pragma unroll
        for (int r = 0; r < 8; ++r) vpack[r] = (_Float16)(acc[2][nt][r] + bvv);
        const int grow = rowbase + 8 * kh2;          // 64-row blocks never straddle batch
        const int bb = grow >> 11, s0 = grow & (SEQ - 1);
        *(v8h*)(vt + ((size_t)bb * HEAD + nt * 16 + m) * SEQ + s0) = vpack;
    }
}

// ---------------------------------------------------------------------------
// Kernel 2: causal flash attention. Block = 4 waves = 64 query rows sharing
// TDM-staged K/V chunks (K: contiguous 4KB 1D tile; V: 2D 64x64B tile with
// stride S*2B, compacted to [64][32] in LDS), double-buffered one chunk ahead.
// Uniform per-block kv loop bound (legal __syncthreads); fully-masked chunks
// contribute exactly zero (mask value -1e30, never -inf -> no NaN paths).
// ---------------------------------------------------------------------------
__global__ __launch_bounds__(128, 1) void attn_kernel(
    const _Float16* __restrict__ qh, const _Float16* __restrict__ kh,
    const _Float16* __restrict__ vt, float* __restrict__ out) {
    __shared__ __align__(32) _Float16 kbuf[2][32 * HEAD];   // 8KB
    __shared__ __align__(32) _Float16 vbuf[2][HEAD * 32];   // 8KB
    __shared__ __align__(32) _Float16 pstage[4][16][32];    // 4KB (per-wave)
    const int lane = threadIdx.x & 31;
    const int wave = threadIdx.x >> 5;
    const int m    = lane & 15;
    const int kh2  = lane >> 4;
    const int b    = blockIdx.y;
    const int qblk = blockIdx.x;
    const int qbase = qblk * 64 + wave * 16;   // within batch
    const size_t bQ = (size_t)b * SEQ * HEAD;
    const size_t bV = (size_t)b * HEAD * SEQ;

    // Q A-fragments, two 32-deep chunks of the 64-dim head
    F16x16 qa[2];
#pragma unroll
    for (int c = 0; c < 2; ++c) {
        const _Float16* p = qh + bQ + (size_t)(qbase + m) * HEAD + c * 32 + kh2 * 16;
        qa[c].h[0] = *(const v8h*)p;
        qa[c].h[1] = *(const v8h*)(p + 8);
    }

    const v8f zero = {0.f,0.f,0.f,0.f,0.f,0.f,0.f,0.f};
    v8f o[4];
    float mrow[8], lrow[8];
#pragma unroll
    for (int nt = 0; nt < 4; ++nt) o[nt] = zero;
#pragma unroll
    for (int r = 0; r < 8; ++r) { mrow[r] = -1e30f; lrow[r] = 0.f; }

    const int kvend = qblk * 64 + 64;          // uniform across the block

    // ---- prologue: TDM-stage KV chunk 0 ----
    if (wave == 0) {
        tdm_load_2d(&kbuf[0][0], kh + bQ, 32 * HEAD / 2, 1, 32 * HEAD / 2);
        tdm_load_2d(&vbuf[0][0], vt + bV, 16, HEAD, SEQ / 2);
    }
    __builtin_amdgcn_s_wait_tensorcnt(0);
    __syncthreads();

    for (int kc = 0; kc < kvend; kc += 32) {
        const int cur = (kc >> 5) & 1;
        if (wave == 0 && kc + 32 < kvend) {    // prefetch next KV chunk via TDM
            tdm_load_2d(&kbuf[cur ^ 1][0], kh + bQ + (size_t)(kc + 32) * HEAD,
                        32 * HEAD / 2, 1, 32 * HEAD / 2);
            tdm_load_2d(&vbuf[cur ^ 1][0], vt + bV + kc + 32, 16, HEAD, SEQ / 2);
        }

        // ---- scores S[16x32] = (Q*8) @ K^T : batch K frags, then WMMAs ----
        v16h kb[2][2];
#pragma unroll
        for (int j = 0; j < 2; ++j)
#pragma unroll
            for (int c = 0; c < 2; ++c)
                kb[j][c] = *(const v16h*)
                    &kbuf[cur][(j * 16 + m) * HEAD + c * 32 + kh2 * 16];
        v8f s[2];
#pragma unroll
        for (int j = 0; j < 2; ++j) {
            s[j] = zero;
#pragma unroll
            for (int c = 0; c < 2; ++c)
                s[j] = __builtin_amdgcn_wmma_f32_16x16x32_f16(
                    false, qa[c].v, false, kb[j][c], (short)0, s[j], false, false);
        }
        // ---- causal mask ----
#pragma unroll
        for (int j = 0; j < 2; ++j) {
            const int kcol = kc + j * 16 + m;
#pragma unroll
            for (int r = 0; r < 8; ++r)
                if (kcol > qbase + r + 8 * kh2) s[j][r] = -1e30f;
        }
        // ---- online softmax: row max / rescale / exp / row sum ----
        float mnew[8], alpha[8];
#pragma unroll
        for (int r = 0; r < 8; ++r) {
            float v = fmaxf(s[0][r], s[1][r]);
#pragma unroll
            for (int sh = 1; sh < 16; sh <<= 1) v = fmaxf(v, __shfl_xor(v, sh, 32));
            mnew[r]  = fmaxf(mrow[r], v);
            alpha[r] = __expf(mrow[r] - mnew[r]);
            mrow[r]  = mnew[r];
        }
#pragma unroll
        for (int r = 0; r < 8; ++r) {
            s[0][r] = __expf(s[0][r] - mnew[r]);
            s[1][r] = __expf(s[1][r] - mnew[r]);
            float v = s[0][r] + s[1][r];
#pragma unroll
            for (int sh = 1; sh < 16; sh <<= 1) v += __shfl_xor(v, sh, 32);
            lrow[r] = lrow[r] * alpha[r] + v;
        }
#pragma unroll
        for (int nt = 0; nt < 4; ++nt)
#pragma unroll
            for (int r = 0; r < 8; ++r) o[nt][r] *= alpha[r];

        // ---- P: C layout -> f16 A layout via per-wave LDS tile ----
        __syncthreads();                        // WAR vs previous iter's reads
#pragma unroll
        for (int j = 0; j < 2; ++j)
#pragma unroll
            for (int r = 0; r < 8; ++r)
                pstage[wave][r + 8 * kh2][j * 16 + m] = (_Float16)s[j][r];
        __syncthreads();                        // RAW before fragment reads
        F16x16 pa;
        pa.h[0] = *(const v8h*)&pstage[wave][m][kh2 * 16];
        pa.h[1] = *(const v8h*)&pstage[wave][m][kh2 * 16 + 8];

        // ---- O += P @ V : batch V frags from staged [64][32] tile ----
        v16h vb[4];
#pragma unroll
        for (int nt = 0; nt < 4; ++nt)
            vb[nt] = *(const v16h*)&vbuf[cur][(nt * 16 + m) * 32 + kh2 * 16];
#pragma unroll
        for (int nt = 0; nt < 4; ++nt)
            o[nt] = __builtin_amdgcn_wmma_f32_16x16x32_f16(
                false, pa.v, false, vb[nt], (short)0, o[nt], false, false);

        __builtin_amdgcn_s_wait_tensorcnt(0);
        __syncthreads();                        // publish next KV buffers
    }

    // ---- epilogue: divide by row sum, store fp32 [B,S,64] ----
    float inv[8];
#pragma unroll
    for (int r = 0; r < 8; ++r) inv[r] = 1.0f / lrow[r];
#pragma unroll
    for (int nt = 0; nt < 4; ++nt)
#pragma unroll
        for (int r = 0; r < 8; ++r)
            out[((size_t)b * SEQ + qbase + r + 8 * kh2) * HEAD + nt * 16 + m] =
                o[nt][r] * inv[r];
}

// ---------------------------------------------------------------------------
extern "C" void kernel_launch(void* const* d_in, const int* in_sizes, int n_in,
                              void* d_out, int out_size, void* d_ws, size_t ws_size,
                              hipStream_t stream) {
    (void)in_sizes; (void)n_in; (void)out_size; (void)ws_size;
    const float* x  = (const float*)d_in[0];
    const float* wq = (const float*)d_in[1];
    const float* bq = (const float*)d_in[2];
    const float* wk = (const float*)d_in[3];
    const float* bk = (const float*)d_in[4];
    const float* wv = (const float*)d_in[5];
    const float* bv = (const float*)d_in[6];

    _Float16* wsH = (_Float16*)d_ws;           // needs ~3.4 MB of d_ws
    _Float16* wf  = wsH;
    _Float16* qh  = wsH + QH_OFF;
    _Float16* khp = wsH + KH_OFF;
    _Float16* vtp = wsH + VT_OFF;

    prep_w_kernel<<<(WF_HALFS + 255) / 256, 256, 0, stream>>>(wq, wk, wv, wf);
    qkv_kernel<<<NROWS / 64, 128, 0, stream>>>(x, bq, bk, bv, wf, qh, khp, vtp);
    attn_kernel<<<dim3(SEQ / 64, BATCH), 128, 0, stream>>>(qh, khp, vtp, (float*)d_out);
}
